// EpisodicMemory_70884140253775
// MI455X (gfx1250) — matrix operand order repaired and checked by reading, compiled, not run on hardware
//
#include <hip/hip_runtime.h>

// ---------------------------------------------------------------------------
// EpisodicMemory forward for MI455X (gfx1250, wave32, WMMA + TDM)
// B=8, L=4096, DIM=512 (complex -> 1024 flat), S=32, K=8
// ---------------------------------------------------------------------------

#define B_ 8
#define L_ 4096
#define DIMC 1024          // DIM*2 flattened complex
#define S_ 32
#define TOPK 8

static constexpr float kEps = 1e-8f;

typedef __attribute__((ext_vector_type(16))) __bf16 v16bf;
typedef __attribute__((ext_vector_type(4)))  __bf16 bf16x4;
typedef __attribute__((ext_vector_type(8)))  float  v8f;
typedef __attribute__((ext_vector_type(4)))  unsigned int u32x4;
typedef __attribute__((ext_vector_type(8)))  int i32x8;
typedef __attribute__((ext_vector_type(4)))  int i32x4;

// ---------------------------------------------------------------------------
// Kernel: pack complex weight pair (Wr, Wi) [512x512 each] into the WMMA
// B-operand register image for the real-folded 1024x1024 matrix Wc:
//   Wc[2k  ][2n  ] =  Wr[k][n]   Wc[2k  ][2n+1] =  Wi[k][n]
//   Wc[2k+1][2n  ] = -Wi[k][n]   Wc[2k+1][2n+1] =  Wr[k][n]
// Packed layout: [n_tile(64)][k_step(32)][lane(32)][16 bf16].
// ---------------------------------------------------------------------------
__global__ __launch_bounds__(256) void pack_w_kernel(
    const float* __restrict__ Wr, const float* __restrict__ Wi,
    __bf16* __restrict__ Wp)
{
  int idx = blockIdx.x * 256 + threadIdx.x;      // 0 .. 1M-1
  int j    = idx & 15;
  int lane = (idx >> 4) & 31;
  int ks   = (idx >> 9) & 31;
  int nt   = idx >> 14;                          // 0..63
  int N = nt * 16 + (lane & 15);
  int K = ks * 32 + ((lane >> 4) ? 16 : 0) + j;
  int k = K >> 1, n = N >> 1;
  float w;
  if ((K & 1) == 0) w = ((N & 1) == 0) ?  Wr[k * 512 + n] : Wi[k * 512 + n];
  else              w = ((N & 1) == 0) ? -Wi[k * 512 + n] : Wr[k * 512 + n];
  Wp[idx] = (__bf16)w;
}

// ---------------------------------------------------------------------------
// Kernel: per-token phase (complex matvec vs W_sal) + average |z|
// ---------------------------------------------------------------------------
__global__ __launch_bounds__(256) void phase_mag_kernel(
    const float* __restrict__ z, const float* __restrict__ Wsr,
    const float* __restrict__ Wsi, float* __restrict__ phase,
    float* __restrict__ avgmag)
{
  int lane = threadIdx.x & 31, wave = threadIdx.x >> 5;
  int tok = blockIdx.x * 8 + wave;               // < B*L
  const float* zp = z + (size_t)tok * DIMC;
  float pr = 0.f, pi = 0.f, ms = 0.f;
  for (int i = lane; i < 512; i += 32) {
    float zr = zp[2 * i], zi = zp[2 * i + 1];
    float wr = Wsr[i], wi = Wsi[i];
    pr += zr * wr - zi * wi;
    pi += zr * wi + zi * wr;
    ms += sqrtf(zr * zr + zi * zi + kEps);
  }
  for (int off = 16; off; off >>= 1) {
    pr += __shfl_xor(pr, off);
    pi += __shfl_xor(pi, off);
    ms += __shfl_xor(ms, off);
  }
  if (lane == 0) {
    phase[tok] = sqrtf(pr * pr + pi * pi + kEps);
    avgmag[tok] = ms * (1.f / 512.f);
  }
}

// ---------------------------------------------------------------------------
// Kernel: salience = sigmoid(phase + novelty + bias); above = salience > 0.5
// ---------------------------------------------------------------------------
__global__ __launch_bounds__(256) void salience_kernel(
    const float* __restrict__ phase, const float* __restrict__ avgmag,
    const float* __restrict__ bias, const float* __restrict__ nscale,
    float* __restrict__ sal_out, int* __restrict__ above)
{
  int idx = blockIdx.x * 256 + threadIdx.x;      // < B*L
  int b = idx >> 12, l = idx & (L_ - 1);
  const float* am = avgmag + (size_t)b * L_;
  float s = 0.f;
  #pragma unroll
  for (int o = -2; o <= 2; ++o) {
    int ll = l + o;
    s += (ll >= 0 && ll < L_) ? am[ll] : 0.f;
  }
  float local = s * 0.2f;
  float nov = (am[l] - local) * nscale[0];
  float x = phase[idx] + nov + bias[0];
  float sv = 1.f / (1.f + __expf(-x));
  sal_out[idx] = sv;
  above[idx] = (sv > 0.5f) ? 1 : 0;
}

// ---------------------------------------------------------------------------
// Kernel: find contiguous runs of `above` per batch (events).
// ---------------------------------------------------------------------------
__global__ __launch_bounds__(256) void run_scan_kernel(
    const int* __restrict__ above, int* __restrict__ run_start,
    int* __restrict__ run_end)
{
  int b = blockIdx.x;
  const int* ab = above + (size_t)b * L_;
  int tid = threadIdx.x;
  if (tid < S_) { run_start[b * S_ + tid] = 0; run_end[b * S_ + tid] = -1; }
  __shared__ int scan[256];
  int base = tid * 16;
  int cnt = 0;
  for (int i = 0; i < 16; ++i) {
    int l = base + i;
    int a = ab[l];
    int p = (l > 0) ? ab[l - 1] : 0;
    cnt += (a && !p) ? 1 : 0;
  }
  scan[tid] = cnt;
  __syncthreads();
  for (int off = 1; off < 256; off <<= 1) {
    int v = (tid >= off) ? scan[tid - off] : 0;
    __syncthreads();
    scan[tid] += v;
    __syncthreads();
  }
  int excl = (tid > 0) ? scan[tid - 1] : 0;
  int rid = excl - 1;
  for (int i = 0; i < 16; ++i) {
    int l = base + i;
    int a = ab[l];
    int p = (l > 0) ? ab[l - 1] : 0;
    int nx = (l < L_ - 1) ? ab[l + 1] : 0;
    if (a && !p) rid++;
    if (a && rid >= 0 && rid < S_) {
      if (!p)  run_start[b * S_ + rid] = l;
      if (!nx) run_end[b * S_ + rid] = l;
    }
  }
}

// ---------------------------------------------------------------------------
// Kernel: salience-weighted mean of z over each run (deterministic, no
// atomics); event_mask; new_mask = min(slot_mask + event_mask, 1).
// ---------------------------------------------------------------------------
__global__ __launch_bounds__(256) void event_accum_kernel(
    const float* __restrict__ z, const float* __restrict__ sal,
    const int* __restrict__ run_start, const int* __restrict__ run_end,
    float* __restrict__ event_vecs, float* __restrict__ event_mask,
    const float* __restrict__ slot_mask, float* __restrict__ new_mask)
{
  int bs = blockIdx.x;
  int b = bs >> 5;
  int st = run_start[bs], en = run_end[bs];
  int tid = threadIdx.x;
  __shared__ float red[256];
  float dden = 0.f;
  if (en >= st)
    for (int l = st + tid; l <= en; l += 256) dden += sal[(size_t)b * L_ + l];
  red[tid] = dden;
  __syncthreads();
  for (int o = 128; o; o >>= 1) {
    if (tid < o) red[tid] += red[tid + o];
    __syncthreads();
  }
  float den = red[0];
  float scale = 1.f / fmaxf(den, kEps);
  float acc0 = 0.f, acc1 = 0.f, acc2 = 0.f, acc3 = 0.f;
  if (en >= st) {
    for (int l = st; l <= en; ++l) {
      float sw = sal[(size_t)b * L_ + l];
      const float* zp = z + ((size_t)b * L_ + l) * DIMC;
      acc0 += sw * zp[tid];
      acc1 += sw * zp[tid + 256];
      acc2 += sw * zp[tid + 512];
      acc3 += sw * zp[tid + 768];
    }
  }
  float* ev = event_vecs + (size_t)bs * DIMC;
  ev[tid]       = acc0 * scale;
  ev[tid + 256] = acc1 * scale;
  ev[tid + 512] = acc2 * scale;
  ev[tid + 768] = acc3 * scale;
  if (tid == 0) {
    float em = (den > 0.f) ? 1.f : 0.f;
    event_mask[bs] = em;
    new_mask[bs] = fminf(slot_mask[bs] + em, 1.f);
  }
}

// ---------------------------------------------------------------------------
// WMMA bf16 GEMM: C(M x 1024) = A(M x 1024, f32 -> bf16) @ Wc(1024 x 1024).
// Block = 8 waves covering M=16, N=1024; each wave owns a 16x128 tile
// (8 n-tiles of 16x16 f32 accumulators = 64 acc VGPRs).
// A tile DMA'd into LDS by the Tensor Data Mover, converted once into the
// bf16 A-fragment image. K loop = 1 ds_load fragment + 8 B fragments issued
// as ONE clause (sched_barrier keeps all 16 b128 loads grouped so their L2
// latency overlaps a single wait) + 8 v_wmma. B addressing strength-reduced:
// one pointer += 1024 B per k-step, nt via constant instruction offsets.
// Dynamic LDS: 65536 B f32 tile + 32768 B bf16 image = 98304 B (3 blocks/WGP).
// ---------------------------------------------------------------------------
template <int STORE_BF16>
__global__ __launch_bounds__(256) void wmma_gemm_kernel(
    const float* __restrict__ A, const __bf16* __restrict__ Wp,
    float* __restrict__ Cf, __bf16* __restrict__ Cb)
{
  extern __shared__ char smem[];
  float* smemA = (float*)smem;                   // 16 x 1024 f32 (64 KiB)
  __bf16* smemBf = (__bf16*)(smem + 65536);      // [ks(32)][lane(32)][16]

  const int tid = threadIdx.x;
  const int lane = tid & 31;
  const int wave = tid >> 5;
  const int hl = lane >> 4;
  const int lm = lane & 15;
  const int m0 = blockIdx.x * 16;
  const int nt0 = wave * 8;                      // n-tile base (16 cols/tile)

  const float* gA = A + (size_t)m0 * DIMC;

#if defined(__gfx1250__) && __has_builtin(__builtin_amdgcn_tensor_load_to_lds)
  // --- Tensor Data Mover: DMA the 16x1024 f32 tile into LDS ---------------
  if (wave == 0) {
    unsigned long long ga = (unsigned long long)(size_t)gA;
    unsigned lds = (unsigned)(size_t)smemA;
    u32x4 g0 = {0u, 0u, 0u, 0u};
    g0[0] = 1u;                                        // count=1, user mode
    g0[1] = lds;                                       // lds_addr
    g0[2] = (unsigned)(ga & 0xffffffffull);            // global_addr[31:0]
    g0[3] = (unsigned)((ga >> 32) & 0x1ffffffull) | (2u << 30); // hi + type=2
    i32x8 g1 = {0, 0, 0, 0, 0, 0, 0, 0};
    g1[0] = (int)(2u << 16);                           // data_size = 4 bytes
    g1[1] = (int)(1024u << 16);                        // tensor_dim0 @bit48
    g1[2] = (int)(16u << 16);                          // tensor_dim1 @bit80
    g1[3] = (int)(1024u << 16);                        // tile_dim0 @bit112
    g1[4] = 16;                                        // tile_dim1 @bit128
    g1[5] = 1024;                                      // dim0_stride @bit160
    i32x4 g2 = {0, 0, 0, 0};
    i32x4 g3 = {0, 0, 0, 0};
#if __clang_major__ >= 23
    i32x8 g4 = {0, 0, 0, 0, 0, 0, 0, 0};
    __builtin_amdgcn_tensor_load_to_lds(g0, g1, g2, g3, g4, 0);
#else
    __builtin_amdgcn_tensor_load_to_lds(g0, g1, g2, g3, 0);
#endif
#if __has_builtin(__builtin_amdgcn_s_wait_tensorcnt)
    __builtin_amdgcn_s_wait_tensorcnt(0);
#endif
  }
  __syncthreads();
#else
  // --- fallback: cooperative staging ---------------------------------------
  for (int i = tid; i < 16 * DIMC / 4; i += 256)
    ((float4*)smemA)[i] = ((const float4*)gA)[i];
  __syncthreads();
#endif

  // --- single-pass f32 -> bf16 conversion into the A-fragment image -------
  for (int e = tid; e < 32 * 32; e += 256) {     // one 16-elem fragment each
    int flane = e & 31;
    int fks = e >> 5;
    int frow = flane & 15;
    int kbase = fks * 32 + ((flane >> 4) ? 16 : 0);
    const float* src = smemA + frow * DIMC + kbase;
    __bf16* dst = smemBf + e * 16;
    #pragma unroll
    for (int j = 0; j < 8; ++j) dst[j] = (__bf16)src[j];
    #pragma unroll
    for (int j = 0; j < 8; ++j) dst[8 + j] = (__bf16)src[8 + j];
  }
  __syncthreads();

  v8f zero8 = {0.f, 0.f, 0.f, 0.f, 0.f, 0.f, 0.f, 0.f};
  v8f acc[8];
  #pragma unroll
  for (int j = 0; j < 8; ++j) acc[j] = zero8;

  // loop-invariant bases; per-k-step increments are constant
  const __bf16* bks = Wp + (size_t)nt0 * 16384 + (size_t)lane * 16;
  const __bf16* aks = smemBf + (size_t)lane * 16;

  #pragma unroll 1
  for (int ks = 0; ks < 32; ++ks) {
    v16bf afr = *(const v16bf*)aks;
    v16bf bfr[8];
    #pragma unroll
    for (int nt = 0; nt < 8; ++nt)
      bfr[nt] = *(const v16bf*)(bks + nt * 16384);   // +nt*32768 B imm offset
#if __has_builtin(__builtin_amdgcn_sched_barrier)
    // keep all 16 b128 loads grouped ahead of the WMMA block so their L2
    // latency is paid once, not 8 times
    __builtin_amdgcn_sched_barrier(0);
#endif
    #pragma unroll
    for (int nt = 0; nt < 8; ++nt)
      acc[nt] = __builtin_amdgcn_wmma_f32_16x16x32_bf16(
          false, afr, false, bfr[nt], (short)0, acc[nt], false, false);
    bks += 512;                                      // next k-step (+1024 B)
    aks += 512;                                      // next A fragment row
  }

  // C/D layout: VGPR r -> M = r (lanes 0-15) / M = 8+r (lanes 16-31), N = lm.
  #pragma unroll
  for (int nt = 0; nt < 8; ++nt)
    #pragma unroll
    for (int r = 0; r < 8; ++r) {
      int row = m0 + (hl ? 8 + r : r);
      int col = (nt0 + nt) * 16 + lm;
      float v = acc[nt][r];
      if (STORE_BF16) Cb[(size_t)row * DIMC + col] = (__bf16)v;
      else            Cf[(size_t)row * DIMC + col] = v;
    }
}

// ---------------------------------------------------------------------------
// Kernel: blend event projection with prior slots, optional key-magnitude.
// ---------------------------------------------------------------------------
__global__ __launch_bounds__(256) void blend_kernel(
    const float* __restrict__ raw, const float* __restrict__ slot,
    const float* __restrict__ event_mask, float* __restrict__ outp,
    float* __restrict__ kmag)
{
  int bs = blockIdx.x, tid = threadIdx.x;
  float m = event_mask[bs];
  float ss = 0.f;
  #pragma unroll
  for (int c = 0; c < 4; ++c) {
    int j = tid + 256 * c;
    size_t o = (size_t)bs * DIMC + j;
    float v = m * raw[o] + (1.f - m) * slot[o];
    outp[o] = v;
    ss += v * v;
  }
  if (kmag) {
    __shared__ float red[256];
    red[tid] = ss;
    __syncthreads();
    for (int o = 128; o; o >>= 1) {
      if (tid < o) red[tid] += red[tid + o];
      __syncthreads();
    }
    if (tid == 0) kmag[bs] = sqrtf(red[0] + kEps);
  }
}

// ---------------------------------------------------------------------------
// Kernel: scores + top-8 + softmax + retrieval + RMS norm. One wave per token.
// lane = slot for the dot phase; lane = dim-group for the gather phase.
// ---------------------------------------------------------------------------
__global__ __launch_bounds__(256) void attention_kernel(
    const __bf16* __restrict__ q, const float* __restrict__ keys,
    const float* __restrict__ values, const float* __restrict__ kmag,
    const float* __restrict__ mask, const float* __restrict__ gain,
    float* __restrict__ out)
{
  int lane = threadIdx.x & 31, wave = threadIdx.x >> 5;
  int tok = blockIdx.x * 8 + wave;               // flat b*L + l
  int b = tok >> 12;
  const __bf16* qp = q + (size_t)tok * DIMC;
  const float* kp = keys + ((size_t)b * S_ + lane) * DIMC;

  float dot = 0.f, qsq = 0.f;
  for (int d = 0; d < DIMC; d += 4) {
    bf16x4 qv = *(const bf16x4*)(qp + d);
    float4 kv = *(const float4*)(kp + d);
    float q0 = (float)qv[0], q1 = (float)qv[1];
    float q2 = (float)qv[2], q3 = (float)qv[3];
    dot += q0 * kv.x + q1 * kv.y + q2 * kv.z + q3 * kv.w;
    qsq += q0 * q0 + q1 * q1 + q2 * q2 + q3 * q3;
  }
  float qmag = sqrtf(qsq + kEps);
  float sc = dot / (qmag * kmag[b * S_ + lane] + kEps);
  if (mask[b * S_ + lane] == 0.f) sc = -1e9f;

  // top-8 across the 32 lanes (ties -> lowest index, like lax.top_k)
  float work = sc;
  float selv[TOPK];
  int seli[TOPK];
  #pragma unroll
  for (int t = 0; t < TOPK; ++t) {
    float v = work;
    int idx = lane;
    for (int off = 16; off; off >>= 1) {
      float ov = __shfl_xor(v, off);
      int oi = __shfl_xor(idx, off);
      if (ov > v || (ov == v && oi < idx)) { v = ov; idx = oi; }
    }
    selv[t] = v;
    seli[t] = idx;
    if (lane == idx) work = -INFINITY;
  }
  float m0 = selv[0];
  float w[TOPK];
  float sum = 0.f;
  #pragma unroll
  for (int t = 0; t < TOPK; ++t) { w[t] = __expf(selv[t] - m0); sum += w[t]; }
  float winv = 1.f / sum;
  #pragma unroll
  for (int t = 0; t < TOPK; ++t) w[t] *= winv;

  // prefetch the 8 selected value rows (L2-resident), then weighted gather
  const float* vb = values + (size_t)b * S_ * DIMC;
  #pragma unroll
  for (int t = 0; t < TOPK; ++t)
    __builtin_prefetch(vb + (size_t)seli[t] * DIMC + lane * 32, 0, 2);

  float vals[32];
  float ssq = 0.f;
  #pragma unroll 4
  for (int i = 0; i < 32; ++i) {
    int j = i * 32 + lane;
    float v = 0.f;
    #pragma unroll
    for (int t = 0; t < TOPK; ++t) v += w[t] * vb[(size_t)seli[t] * DIMC + j];
    vals[i] = v;
    ssq += v * v;
  }
  for (int off = 16; off; off >>= 1) ssq += __shfl_xor(ssq, off);
  float rinv = 1.f / sqrtf(ssq * (1.f / 512.f) + kEps);
  float* op = out + (size_t)tok * DIMC;
  #pragma unroll 4
  for (int i = 0; i < 32; ++i) {
    int j = i * 32 + lane;
    op[j] = vals[i] * rinv * gain[j >> 1];
  }
}

// ---------------------------------------------------------------------------
// Launch
// ---------------------------------------------------------------------------
extern "C" void kernel_launch(void* const* d_in, const int* in_sizes, int n_in,
                              void* d_out, int out_size, void* d_ws,
                              size_t ws_size, hipStream_t stream)
{
  (void)in_sizes; (void)n_in; (void)out_size; (void)ws_size;
  const float* z           = (const float*)d_in[0];
  const float* slot_keys   = (const float*)d_in[1];
  const float* slot_values = (const float*)d_in[2];
  const float* slot_mask   = (const float*)d_in[3];
  const float* W_sal_r     = (const float*)d_in[4];
  const float* W_sal_i     = (const float*)d_in[5];
  const float* score_bias  = (const float*)d_in[6];
  const float* nov_scale   = (const float*)d_in[7];
  const float* W_ek_r      = (const float*)d_in[8];
  const float* W_ek_i      = (const float*)d_in[9];
  const float* W_ev_r      = (const float*)d_in[10];
  const float* W_ev_i      = (const float*)d_in[11];
  const float* W_rq_r      = (const float*)d_in[12];
  const float* W_rq_i      = (const float*)d_in[13];
  const float* norm_gain   = (const float*)d_in[14];

  // output layout (flat f32): out | new_keys | new_values | new_mask | salience
  float* out_main   = (float*)d_out;
  float* out_nk     = out_main + (size_t)B_ * L_ * DIMC;     // 33,554,432
  float* out_nv     = out_nk + (size_t)B_ * S_ * DIMC;       // +262,144
  float* out_nm     = out_nv + (size_t)B_ * S_ * DIMC;       // +262,144
  float* out_sal    = out_nm + B_ * S_;                      // +256

  // workspace layout (bytes)
  char* ws = (char*)d_ws;
  __bf16* wp_ek   = (__bf16*)(ws + 0);                       // 2 MiB
  __bf16* wp_ev   = (__bf16*)(ws + 2097152);                 // 2 MiB
  __bf16* wp_rq   = (__bf16*)(ws + 4194304);                 // 2 MiB
  float*  phase   = (float*)(ws + 6291456);                  // 128 KiB
  float*  avgm    = (float*)(ws + 6422528);                  // 128 KiB
  int*    above   = (int*)(ws + 6553600);                    // 128 KiB
  int*    rstart  = (int*)(ws + 6684672);                    // 1 KiB
  int*    rend    = (int*)(ws + 6685696);                    // 1 KiB
  float*  evvec   = (float*)(ws + 6686720);                  // 1 MiB
  float*  emask   = (float*)(ws + 7735296);                  // 1 KiB
  float*  kmag    = (float*)(ws + 7736320);                  // 1 KiB
  float*  ek_raw  = (float*)(ws + 7737344);                  // 1 MiB
  float*  ev_raw  = (float*)(ws + 8785920);                  // 1 MiB
  __bf16* q_bf    = (__bf16*)(ws + 9834496);                 // 64 MiB

  const size_t gemm_lds = 65536 + 32768;                     // 96 KiB

  // 1) pack complex weights into WMMA B-fragment images
  pack_w_kernel<<<4096, 256, 0, stream>>>(W_ek_r, W_ek_i, wp_ek);
  pack_w_kernel<<<4096, 256, 0, stream>>>(W_ev_r, W_ev_i, wp_ev);
  pack_w_kernel<<<4096, 256, 0, stream>>>(W_rq_r, W_rq_i, wp_rq);

  // 2) salience pipeline
  phase_mag_kernel<<<4096, 256, 0, stream>>>(z, W_sal_r, W_sal_i, phase, avgm);
  salience_kernel<<<128, 256, 0, stream>>>(phase, avgm, score_bias, nov_scale,
                                           out_sal, above);
  run_scan_kernel<<<B_, 256, 0, stream>>>(above, rstart, rend);
  event_accum_kernel<<<B_ * S_, 256, 0, stream>>>(
      z, out_sal, rstart, rend, evvec, emask, slot_mask, out_nm);

  // 3) event key/value projection (WMMA + TDM) + blend with prior slots
  wmma_gemm_kernel<0><<<(B_ * S_) / 16, 256, gemm_lds, stream>>>(
      evvec, wp_ek, ek_raw, nullptr);
  wmma_gemm_kernel<0><<<(B_ * S_) / 16, 256, gemm_lds, stream>>>(
      evvec, wp_ev, ev_raw, nullptr);
  blend_kernel<<<B_ * S_, 256, 0, stream>>>(ek_raw, slot_keys, emask, out_nk,
                                            kmag);
  blend_kernel<<<B_ * S_, 256, 0, stream>>>(ev_raw, slot_values, emask, out_nv,
                                            nullptr);

  // 4) query projection: the 68.7 GFLOP GEMM (WMMA bf16, TDM-staged A)
  wmma_gemm_kernel<1><<<(B_ * L_) / 16, 256, gemm_lds, stream>>>(
      z, wp_rq, nullptr, q_bf);

  // 5) scores + top-k + softmax + retrieval + RMS norm
  attention_kernel<<<(B_ * L_) / 8, 256, 0, stream>>>(
      q_bf, out_nk, out_nv, kmag, out_nm, norm_gain, out_main);
}
